// MultiHeadSelfAttention_71442486002121
// MI455X (gfx1250) — matrix operand order repaired and checked by reading, not compiled, not run on hardware
//
#include <hip/hip_runtime.h>
#include <hip/hip_bf16.h>

// ---------------------------------------------------------------------------
// MI455X (gfx1250) multi-head self-attention, bf16 WMMA + TDM-staged GEMMs.
//   x[4,2048,1024] -> LN -> Q/K/V -> flash softmax(QK^T/8)V -> out proj
// ---------------------------------------------------------------------------

typedef __bf16 bf16_t;
typedef __bf16 v16bf __attribute__((ext_vector_type(16)));
typedef float  v8f   __attribute__((ext_vector_type(8)));
typedef unsigned int u32x4 __attribute__((ext_vector_type(4)));
typedef int          i32x4v __attribute__((ext_vector_type(4)));
typedef int          i32x8v __attribute__((ext_vector_type(8)));

#define N_FEAT 1024
#define N_HEAD 16
#define D_KH   64
#define SEQ    2048
#define BATCH  4
#define ROWS   (BATCH * SEQ)    // 8192

// GEMM tiling: 4 waves/block, each wave 64x64 out, block 256x64.
#define GEMM_WAVES 4
#define WAVE_M     64
#define BLOCK_M    (GEMM_WAVES * WAVE_M)   // 256
#define TILE_N     64
#define KSTEP      32
// LDS B-tile row stride: 32 bf16 (64B) + 16B pad = 40 elements (80B).
// 80B stride -> 16 distinct banks over 16 lanes, 16B-aligned for b128 LDS ops.
#define BROW_ELEMS 40

#if defined(__has_builtin)
#if __has_builtin(__builtin_amdgcn_tensor_load_to_lds) && __has_builtin(__builtin_amdgcn_s_wait_tensorcnt)
#define HAVE_TDM 1
#endif
#endif
#ifndef HAVE_TDM
#define HAVE_TDM 0
#endif

// fp32 -> bf16, round-to-nearest-even
__device__ __forceinline__ bf16_t f2bf(float x) {
  unsigned u = __float_as_uint(x);
  u += 0x7FFFu + ((u >> 16) & 1u);
  unsigned short hs = (unsigned short)(u >> 16);
  union { unsigned short u; bf16_t b; } cv;
  cv.u = hs;
  return cv.b;
}

union Frag {
  v16bf v;
  uint4 q[2];
};

// A-matrix 16x32 bf16 fragment (ISA 7.12.2): lane&15 = M row, half = lane>>4,
// K = 8*half + {0..7} then 8*half + {16..23}.
__device__ __forceinline__ Frag load_a_frag(const bf16_t* __restrict__ base,
                                            int row_stride, int kbase, int lane) {
  Frag f;
  const bf16_t* p = base + (size_t)(lane & 15) * row_stride + kbase + ((lane >> 4) << 3);
  f.q[0] = *reinterpret_cast<const uint4*>(p);
  f.q[1] = *reinterpret_cast<const uint4*>(p + 16);
  return f;
}

// B-matrix 32x16 fragment from transposed storage Bt[n][k]:
// lane&15 = N col, K = 8*(lane>>4) + {0..7} then 8*(lane>>4) + {16..23}.
__device__ __forceinline__ Frag load_b_frag(const bf16_t* __restrict__ base,
                                            int row_stride, int kbase, int lane) {
  Frag f;
  const bf16_t* p = base + (size_t)(lane & 15) * row_stride + kbase + ((lane >> 4) << 3);
  f.q[0] = *reinterpret_cast<const uint4*>(p);
  f.q[1] = *reinterpret_cast<const uint4*>(p + 16);
  return f;
}

// Same B fragment but from the padded LDS tile (row stride BROW_ELEMS).
__device__ __forceinline__ Frag load_b_frag_lds(const bf16_t* base, int n_base, int lane) {
  Frag f;
  const bf16_t* p = base + (size_t)(n_base + (lane & 15)) * BROW_ELEMS + ((lane >> 4) << 3);
  f.q[0] = *reinterpret_cast<const uint4*>(p);
  f.q[1] = *reinterpret_cast<const uint4*>(p + 16);
  return f;
}

__device__ __forceinline__ v8f wmma_bf16(const Frag& a, const Frag& b, v8f c) {
  return __builtin_amdgcn_wmma_f32_16x16x32_bf16(false, a.v, false, b.v,
                                                 (short)0, c, false, false);
}

#if HAVE_TDM
// Issue a TDM 2D tile load: tile_rows x KSTEP bf16 from row-major global
// (row stride = stride_elems) into LDS at lds_byte_off, with 4-DWORD padding
// after every 16 DWORDs (one 32-elem row) -> LDS row stride BROW_ELEMS.
__device__ __forceinline__ void tdm_load_b_tile(unsigned lds_byte_off,
                                                const bf16_t* gtile,
                                                unsigned tile_rows,
                                                unsigned long long stride_elems) {
  unsigned long long ga = (unsigned long long)(const void*)gtile;
  u32x4 g0;
  g0[0] = 1u;                                            // count=1 (user D#)
  g0[1] = lds_byte_off;                                  // lds_addr (bytes)
  g0[2] = (unsigned)(ga & 0xFFFFFFFFu);                  // global_addr lo
  g0[3] = (unsigned)((ga >> 32) & 0x01FFFFFFu) | (2u << 30);  // addr hi | type=2
  const unsigned td0 = KSTEP;                            // tensor_dim0 (tight)
  const unsigned td1 = tile_rows;                        // tensor_dim1 (tight)
  i32x8v g1;
  g1[0] = (int)((1u << 16)        // data_size = 1 -> 2 bytes
              | (1u << 20)        // pad_enable
              | (3u << 22)        // pad_interval: 16 DWORDs
              | (3u << 25));      // pad_amount: 4 DWORDs (16B)
  g1[1] = (int)((td0 & 0xFFFFu) << 16);                            // dim0[15:0]
  g1[2] = (int)(((td0 >> 16) & 0xFFFFu) | ((td1 & 0xFFFFu) << 16));// dim0 hi|dim1 lo
  g1[3] = (int)(((td1 >> 16) & 0xFFFFu) | ((KSTEP & 0xFFFFu) << 16)); // |tile_dim0
  g1[4] = (int)(tile_rows & 0xFFFFu);                              // tile_dim1; dim2=0
  g1[5] = (int)(unsigned)(stride_elems & 0xFFFFFFFFull);           // dim0_stride lo
  g1[6] = (int)(unsigned)((stride_elems >> 32) & 0xFFFFull);       // dim0_stride hi
  g1[7] = 0;
  i32x4v gz = {};
#if __clang_major__ >= 23
  i32x8v gz8 = {};
  __builtin_amdgcn_tensor_load_to_lds(g0, g1, gz, gz, gz8, 0);
#else
  __builtin_amdgcn_tensor_load_to_lds(g0, g1, gz, gz, 0);
#endif
}
#endif  // HAVE_TDM

// ---------------------------------------------------------------------------
// Kernel 1: LayerNorm(eps=1e-5) + bf16 cast.  One 256-thread block per row.
// ---------------------------------------------------------------------------
__global__ __launch_bounds__(256)
void ln_bf16_kernel(const float* __restrict__ x, const float* __restrict__ gamma,
                    const float* __restrict__ beta, bf16_t* __restrict__ xn) {
  const int row = blockIdx.x;
  const int tid = threadIdx.x;
  const float* xr = x + (size_t)row * N_FEAT;
  float4 v = reinterpret_cast<const float4*>(xr)[tid];
  float a[4] = {v.x, v.y, v.z, v.w};
  float s = 0.f, ss = 0.f;
#pragma unroll
  for (int j = 0; j < 4; ++j) { s += a[j]; ss += a[j] * a[j]; }
#pragma unroll
  for (int off = 16; off > 0; off >>= 1) {
    s  += __shfl_xor(s,  off);
    ss += __shfl_xor(ss, off);
  }
  __shared__ float red[2][8];
  const int wv = tid >> 5, lane = tid & 31;
  if (lane == 0) { red[0][wv] = s; red[1][wv] = ss; }
  __syncthreads();
  if (tid == 0) {
    float aa = 0.f, bb = 0.f;
#pragma unroll
    for (int i = 0; i < 8; ++i) { aa += red[0][i]; bb += red[1][i]; }
    red[0][0] = aa; red[1][0] = bb;
  }
  __syncthreads();
  const float mean = red[0][0] * (1.0f / N_FEAT);
  const float var  = red[1][0] * (1.0f / N_FEAT) - mean * mean;
  const float rstd = rsqrtf(var + 1e-5f);
  bf16_t* outp = xn + (size_t)row * N_FEAT;
  const int i0 = tid * 4;
#pragma unroll
  for (int j = 0; j < 4; ++j) {
    const int i = i0 + j;
    outp[i] = f2bf((a[j] - mean) * rstd * gamma[i] + beta[i]);
  }
}

// ---------------------------------------------------------------------------
// Kernel 2: fp32 -> bf16 weight cast.
// ---------------------------------------------------------------------------
__global__ __launch_bounds__(256)
void cast_bf16_kernel(const float* __restrict__ src, bf16_t* __restrict__ dst, int n) {
  const int i = blockIdx.x * blockDim.x + threadIdx.x;
  if (i < n) dst[i] = f2bf(src[i]);
}

// ---------------------------------------------------------------------------
// Kernel 3: y = A @ W.T + bias.  4 waves/block; block tile 256x64.
//   Wave w owns rows [m0 + w*64, +64), all 64 block cols -> 16 accumulators.
//   B (weight) 64x32 tile staged to LDS via TENSOR_LOAD_TO_LDS, double-buffered;
//   wave 0 issues the DMA + s_wait_tensorcnt, barriers publish to the block.
//   mode: 0 = bf16 [B,H,S,Dk], 1 = bf16 [B,H,Dk,S] (V^T), 2 = fp32 row-major.
// ---------------------------------------------------------------------------
__global__ __launch_bounds__(GEMM_WAVES * 32)
void gemm_kernel(const bf16_t* __restrict__ A, const bf16_t* __restrict__ W,
                 const float* __restrict__ bias, void* __restrict__ dst, int mode) {
  const int tid  = threadIdx.x;
  const int lane = tid & 31;
  const int wv   = tid >> 5;                       // 0..3
  const int m0 = blockIdx.x * BLOCK_M + wv * WAVE_M;
  const int n0 = blockIdx.y * TILE_N;

  __shared__ bf16_t Bl[2][TILE_N * BROW_ELEMS];

  v8f acc[4][4] = {};                              // [row subtile][col subtile]

  const int nk = N_FEAT / KSTEP;

#if HAVE_TDM
  if (wv == 0) {
    tdm_load_b_tile((unsigned)(unsigned long long)(const void*)&Bl[0][0],
                    W + (size_t)n0 * N_FEAT, TILE_N, N_FEAT);
  }
#endif

  for (int i = 0; i < nk; ++i) {
    const int k = i * KSTEP;
    const int cur = i & 1;
#if HAVE_TDM
    if (wv == 0) __builtin_amdgcn_s_wait_tensorcnt(0);   // Bl[cur] ready
    __syncthreads();                                     // publish to block
    if (wv == 0 && i + 1 < nk) {
      tdm_load_b_tile((unsigned)(unsigned long long)(const void*)&Bl[cur ^ 1][0],
                      W + (size_t)n0 * N_FEAT + (k + KSTEP), TILE_N, N_FEAT);
    }
#else
    // Cooperative fallback: 128 threads x 16 bf16 = 64x32 tile.
    {
      const int n = tid >> 1, half = tid & 1;
      const uint4* s4 = reinterpret_cast<const uint4*>(
          W + (size_t)(n0 + n) * N_FEAT + k + half * 16);
      uint4* d4 = reinterpret_cast<uint4*>(&Bl[cur][n * BROW_ELEMS + half * 16]);
      d4[0] = s4[0];
      d4[1] = s4[1];
    }
    __syncthreads();
#endif
    // A fragments: 4 row subtiles, straight from global (L2-resident).
    Frag af[4];
#pragma unroll
    for (int rt = 0; rt < 4; ++rt) {
      const bf16_t* arow = A + (size_t)(m0 + rt * 16) * N_FEAT;
      af[rt] = load_a_frag(arow, N_FEAT, k, lane);
      if (k + KSTEP < N_FEAT)
        __builtin_prefetch(arow + (size_t)(lane & 15) * N_FEAT + k + KSTEP, 0, 1);
    }
    // B fragments from LDS, 16 WMMAs.
#pragma unroll
    for (int ct = 0; ct < 4; ++ct) {
      Frag bfr = load_b_frag_lds(&Bl[cur][0], ct * 16, lane);
#pragma unroll
      for (int rt = 0; rt < 4; ++rt)
        acc[rt][ct] = wmma_bf16(af[rt], bfr, acc[rt][ct]);
    }
    __syncthreads();                                     // reads done before overwrite
  }

  // Epilogue: bias + store per mode.
#pragma unroll
  for (int ct = 0; ct < 4; ++ct) {
    const int n = n0 + ct * 16 + (lane & 15);
    const float bn = bias[n];
    const int h = n >> 6, d = n & (D_KH - 1);
#pragma unroll
    for (int rt = 0; rt < 4; ++rt) {
#pragma unroll
      for (int r = 0; r < 8; ++r) {
        const int m = m0 + rt * 16 + r + ((lane >> 4) << 3);
        const float val = acc[rt][ct][r] + bn;
        if (mode == 2) {
          ((float*)dst)[(size_t)m * N_FEAT + n] = val;
        } else {
          const int b_ = m >> 11, s = m & (SEQ - 1);
          if (mode == 1)
            ((bf16_t*)dst)[((size_t)(b_ * N_HEAD + h) * D_KH + d) * SEQ + s] = f2bf(val);
          else
            ((bf16_t*)dst)[((size_t)(b_ * N_HEAD + h) * SEQ + s) * D_KH + d] = f2bf(val);
        }
      }
    }
  }
}

// ---------------------------------------------------------------------------
// Kernel 4: flash attention.  One wave per (b, h, 16-query tile).
//   32 keys/iter: 4 score WMMAs + online softmax + 4 PV WMMAs; P relayout
//   C->A via a 1 KB per-wave LDS slab (single wave -> no barrier needed).
// ---------------------------------------------------------------------------
__global__ __launch_bounds__(32)
void attn_kernel(const bf16_t* __restrict__ Qb, const bf16_t* __restrict__ Kb,
                 const bf16_t* __restrict__ Vt, bf16_t* __restrict__ Ctx) {
  const int lane = threadIdx.x;
  const int qt = blockIdx.x;            // 0..SEQ/16-1
  const int bh = blockIdx.y;            // b*16+h
  const int b = bh >> 4, h = bh & 15;
  const bf16_t* Qh = Qb + (size_t)bh * SEQ * D_KH;
  const bf16_t* Kh = Kb + (size_t)bh * SEQ * D_KH;
  const bf16_t* Vh = Vt + (size_t)bh * D_KH * SEQ;

  Frag aq[2];
  const bf16_t* qrow = Qh + (size_t)(qt * 16) * D_KH;
  aq[0] = load_a_frag(qrow, D_KH, 0,  lane);
  aq[1] = load_a_frag(qrow, D_KH, 32, lane);

  float m[8], l[8];
  v8f o[4] = {};
#pragma unroll
  for (int r = 0; r < 8; ++r) { m[r] = -3.0e38f; l[r] = 0.0f; }

  __shared__ bf16_t Pl[16 * 32];

  for (int kb = 0; kb < SEQ; kb += 32) {
    v8f sc[2];
#pragma unroll
    for (int j = 0; j < 2; ++j) {
      const bf16_t* krow = Kh + (size_t)(kb + j * 16) * D_KH;
      v8f a = {};
      Frag b0 = load_b_frag(krow, D_KH, 0,  lane);
      a = wmma_bf16(aq[0], b0, a);
      Frag b1 = load_b_frag(krow, D_KH, 32, lane);
      a = wmma_bf16(aq[1], b1, a);
      sc[j] = a;
    }
#pragma unroll
    for (int r = 0; r < 8; ++r) {
      float s0 = sc[0][r] * 0.125f;     // 1/sqrt(64)
      float s1 = sc[1][r] * 0.125f;
      float tmax = fmaxf(s0, s1);
#pragma unroll
      for (int off = 1; off < 16; off <<= 1)
        tmax = fmaxf(tmax, __shfl_xor(tmax, off));
      const float mnew = fmaxf(m[r], tmax);
      const float corr = __expf(m[r] - mnew);
      m[r] = mnew;
      const float p0 = __expf(s0 - mnew);
      const float p1 = __expf(s1 - mnew);
      float ps = p0 + p1;
#pragma unroll
      for (int off = 1; off < 16; off <<= 1) ps += __shfl_xor(ps, off);
      l[r] = l[r] * corr + ps;
#pragma unroll
      for (int nt = 0; nt < 4; ++nt) o[nt][r] *= corr;
      const int prow = r + ((lane >> 4) << 3);
      Pl[prow * 32 + (lane & 15)]      = f2bf(p0);
      Pl[prow * 32 + 16 + (lane & 15)] = f2bf(p1);
    }
    Frag pa;
    {
      const bf16_t* pp = &Pl[(lane & 15) * 32 + ((lane >> 4) << 3)];
      pa.q[0] = *reinterpret_cast<const uint4*>(pp);
      pa.q[1] = *reinterpret_cast<const uint4*>(pp + 16);
    }
#pragma unroll
    for (int nt = 0; nt < 4; ++nt) {
      const bf16_t* vrow = Vh + (size_t)(nt * 16) * SEQ + kb;
      Frag bv = load_b_frag(vrow, SEQ, 0, lane);
      o[nt] = wmma_bf16(pa, bv, o[nt]);
    }
  }

#pragma unroll
  for (int r = 0; r < 8; ++r) {
    const float inv = 1.0f / l[r];
    const int s = qt * 16 + r + ((lane >> 4) << 3);
#pragma unroll
    for (int nt = 0; nt < 4; ++nt) {
      const int col = h * D_KH + nt * 16 + (lane & 15);
      Ctx[(size_t)(b * SEQ + s) * N_FEAT + col] = f2bf(o[nt][r] * inv);
    }
  }
}

// ---------------------------------------------------------------------------
extern "C" void kernel_launch(void* const* d_in, const int* in_sizes, int n_in,
                              void* d_out, int out_size, void* d_ws, size_t ws_size,
                              hipStream_t stream) {
  const float* x  = (const float*)d_in[0];
  const float* lg = (const float*)d_in[1];
  const float* lb = (const float*)d_in[2];
  const float* Wq = (const float*)d_in[3];
  const float* bq = (const float*)d_in[4];
  const float* Wk = (const float*)d_in[5];
  const float* bk = (const float*)d_in[6];
  const float* Wv = (const float*)d_in[7];
  const float* bv = (const float*)d_in[8];
  const float* Wo = (const float*)d_in[9];
  const float* bo = (const float*)d_in[10];
  float* out = (float*)d_out;

  char* ws = (char*)d_ws;
  size_t off = 0;
  auto carve = [&](size_t bytes) {
    char* p = ws + off;
    off += (bytes + 255) & ~(size_t)255;
    return p;
  };
  bf16_t* xn  = (bf16_t*)carve((size_t)ROWS * N_FEAT * 2);
  bf16_t* wqb = (bf16_t*)carve((size_t)N_FEAT * N_FEAT * 2);
  bf16_t* wkb = (bf16_t*)carve((size_t)N_FEAT * N_FEAT * 2);
  bf16_t* wvb = (bf16_t*)carve((size_t)N_FEAT * N_FEAT * 2);
  bf16_t* wob = (bf16_t*)carve((size_t)N_FEAT * N_FEAT * 2);
  bf16_t* Qb  = (bf16_t*)carve((size_t)ROWS * N_FEAT * 2);   // [B,H,S,Dk]
  bf16_t* Kb  = (bf16_t*)carve((size_t)ROWS * N_FEAT * 2);   // [B,H,S,Dk]
  bf16_t* Vt  = (bf16_t*)carve((size_t)ROWS * N_FEAT * 2);   // [B,H,Dk,S]
  bf16_t* Ctx = (bf16_t*)carve((size_t)ROWS * N_FEAT * 2);   // [B,S,F]

  ln_bf16_kernel<<<dim3(ROWS), dim3(256), 0, stream>>>(x, lg, lb, xn);

  const int wn = N_FEAT * N_FEAT;
  const int wblocks = (wn + 255) / 256;
  cast_bf16_kernel<<<dim3(wblocks), dim3(256), 0, stream>>>(Wq, wqb, wn);
  cast_bf16_kernel<<<dim3(wblocks), dim3(256), 0, stream>>>(Wk, wkb, wn);
  cast_bf16_kernel<<<dim3(wblocks), dim3(256), 0, stream>>>(Wv, wvb, wn);
  cast_bf16_kernel<<<dim3(wblocks), dim3(256), 0, stream>>>(Wo, wob, wn);

  dim3 ggrid(ROWS / BLOCK_M, N_FEAT / TILE_N);   // 32 x 16
  dim3 gblock(GEMM_WAVES * 32);
  gemm_kernel<<<ggrid, gblock, 0, stream>>>(xn, wqb, bq, (void*)Qb, 0);
  gemm_kernel<<<ggrid, gblock, 0, stream>>>(xn, wkb, bk, (void*)Kb, 0);
  gemm_kernel<<<ggrid, gblock, 0, stream>>>(xn, wvb, bv, (void*)Vt, 1);

  attn_kernel<<<dim3(SEQ / 16, BATCH * N_HEAD), dim3(32), 0, stream>>>(Qb, Kb, Vt, Ctx);

  gemm_kernel<<<ggrid, gblock, 0, stream>>>(Ctx, wob, bo, (void*)out, 2);

  (void)in_sizes; (void)n_in; (void)out_size; (void)ws_size;
}
